// MoEMLP_tp_6846177870126
// MI455X (gfx1250) — matrix operand order repaired and checked by reading, not compiled, not run on hardware
//
#include <hip/hip_runtime.h>
#include <hip/hip_bf16.h>

// ---------------------------------------------------------------------------
// MoE grouped MLP (fc1 -> gelu -> fc2 -> probs-combine + residual), MI455X.
//
//   Pre-pass:  Xh   = f16(X)                      [T, H]
//              W1t  = f16(W1) transposed          [E, F, H]   (K=H contiguous)
//              W2t  = f16(W2flat) transposed      [H, E*F]    (K=E*F contiguous)
//   Pass 1:    act[t, e*F+f] = f16( probs[t,e] * gelu( Xh[t,:] . W1t[e,f,:] ) )
//   Pass 2:    out[t,h]      = act[t,:] . W2t[h,:] + residual[t,h]
//
// Both GEMMs: 128x128 block tile, 8 wave32s (32x64 wave tile, 2x4 v8f accs),
// K staged 64-wide, DOUBLE-BUFFERED via global_load_async_to_lds_b128 and
// s_wait_asynccnt; MACs via v_wmma_f32_16x16x32_f16.
// probs is already zero off-route, so routing_map (d_in[3]) is unused.
// ---------------------------------------------------------------------------

typedef __attribute__((ext_vector_type(16))) _Float16 v16h;
typedef __attribute__((ext_vector_type(8)))  _Float16 v8h;
typedef __attribute__((ext_vector_type(4)))  _Float16 v4h;
typedef __attribute__((ext_vector_type(8)))  float    v8f;

#define MOE_S 1024
#define MOE_B 2
#define MOE_H 1024
#define MOE_F 4096
#define MOE_E 4
#define MOE_T (MOE_S * MOE_B)      // 2048 tokens
#define MOE_K2 (MOE_E * MOE_F)     // 16384

#define BM 128                     // block tile rows (tokens)
#define BNT 128                    // block tile cols
#define BK 64                      // K per LDS stage (2 WMMA k-steps)
#define LROW 72                    // LDS row stride in halves (144B: 16B-aligned, bank-staggered)
#define STAGE_HALVES (BM * LROW)   // 9216 halves
#define STAGE_BYTES  (STAGE_HALVES * 2)

// --- CDNA5 async copy: 16B global -> LDS, tracked by ASYNCcnt ---------------
__device__ __forceinline__ void async_copy16(unsigned lds_byte_off, const void* gptr) {
    asm volatile("global_load_async_to_lds_b128 %0, %1, off"
                 :: "v"(lds_byte_off), "v"(gptr) : "memory");
}
__device__ __forceinline__ void wait_async_le8() {
    asm volatile("s_wait_asynccnt 0x8" ::: "memory");
}
__device__ __forceinline__ void wait_async_0() {
    asm volatile("s_wait_asynccnt 0x0" ::: "memory");
}

// --- WMMA fragment loads, per CDNA5 ISA 7.12.2 VGPR layouts -----------------
// A (16x32): lane L row M=L&15; halves 0..7 = K[(L>>4)*8..+8), 8..15 = K[16+(L>>4)*8..+8)
__device__ __forceinline__ v16h frag_a(const _Float16* s, int M, int hs, int kk) {
    union { v16h v; v8h h[2]; } f;
    const _Float16* p = s + M * LROW + kk;
    f.h[0] = *(const v8h*)(p + hs * 8);
    f.h[1] = *(const v8h*)(p + 16 + hs * 8);
    return f.v;
}
// B (32x16): lane L col N=L&15; halves 0..7 = K[(L>>4)*8..+8), 8..15 = K[16+(L>>4)*8..+8)
__device__ __forceinline__ v16h frag_b(const _Float16* s, int N, int hs, int kk) {
    union { v16h v; v8h h[2]; } f;
    const _Float16* p = s + N * LROW + kk;
    f.h[0] = *(const v8h*)(p + hs * 8);
    f.h[1] = *(const v8h*)(p + 16 + hs * 8);
    return f.v;
}

__device__ __forceinline__ float gelu_exact(float x) {
    return 0.5f * x * (1.0f + erff(x * 0.70710678118654752f));
}

#define WMMA(A, B, C) __builtin_amdgcn_wmma_f32_16x16x32_f16(false, A, false, B, (short)0, C, false, false)

// Shared double-buffered async GEMM main loop.
// Arows = A + row0*K (M-major, K contiguous); Brows = B + col0*K (N-major, K contiguous).
__device__ __forceinline__ void wmma_mainloop(const _Float16* __restrict__ Arows,
                                              const _Float16* __restrict__ Brows,
                                              int K, _Float16* sA, _Float16* sB,
                                              int tid, int lane, int m0, int n0,
                                              v8f acc[2][4])
{
    const int hs = lane >> 4, la = lane & 15;
    const unsigned aBase = (unsigned)(unsigned long long)(uintptr_t)sA;
    const unsigned bBase = (unsigned)(unsigned long long)(uintptr_t)sB;
    const int nstage = K >> 6;

    auto issue = [&](int s) {
        if (s >= nstage) return;
        const int k0 = s << 6;
        const unsigned ab = aBase + (unsigned)(s & 1) * STAGE_BYTES;
        const unsigned bb = bBase + (unsigned)(s & 1) * STAGE_BYTES;
        #pragma unroll
        for (int i = 0; i < 4; ++i) {                 // 256 rows x 32 halves = 1024x16B chunks
            int c = tid + i * 256, row = c >> 2, sub = c & 3;
            async_copy16(ab + (unsigned)(row * LROW + sub * 8) * 2,
                         Arows + (size_t)row * K + k0 + sub * 8);
        }
        #pragma unroll
        for (int i = 0; i < 4; ++i) {
            int c = tid + i * 256, row = c >> 2, sub = c & 3;
            async_copy16(bb + (unsigned)(row * LROW + sub * 8) * 2,
                         Brows + (size_t)row * K + k0 + sub * 8);
        }
    };

    issue(0);
    issue(1);
    for (int s = 0; s < nstage; ++s) {
        if (s + 1 < nstage) wait_async_le8();   // stage s done; stage s+1 may be in flight
        else                wait_async_0();
        __syncthreads();
        const _Float16* a = sA + (s & 1) * STAGE_HALVES;
        const _Float16* b = sB + (s & 1) * STAGE_HALVES;
        #pragma unroll
        for (int kk = 0; kk < BK; kk += 32) {
            v16h a0 = frag_a(a, m0 + la, hs, kk);
            v16h a1 = frag_a(a, m0 + 16 + la, hs, kk);
            v16h b0 = frag_b(b, n0 + la, hs, kk);
            v16h b1 = frag_b(b, n0 + 16 + la, hs, kk);
            v16h b2 = frag_b(b, n0 + 32 + la, hs, kk);
            v16h b3 = frag_b(b, n0 + 48 + la, hs, kk);
            acc[0][0] = WMMA(a0, b0, acc[0][0]);
            acc[0][1] = WMMA(a0, b1, acc[0][1]);
            acc[0][2] = WMMA(a0, b2, acc[0][2]);
            acc[0][3] = WMMA(a0, b3, acc[0][3]);
            acc[1][0] = WMMA(a1, b0, acc[1][0]);
            acc[1][1] = WMMA(a1, b1, acc[1][1]);
            acc[1][2] = WMMA(a1, b2, acc[1][2]);
            acc[1][3] = WMMA(a1, b3, acc[1][3]);
        }
        __syncthreads();
        issue(s + 2);                            // safe: all waves finished reading buf (s&1)
    }
}

// ---------------------------------------------------------------------------
// Pre-pass A: f32 -> f16 straight conversion (vectorized x4)
// ---------------------------------------------------------------------------
__global__ __launch_bounds__(256)
void cvt_f32_f16(const float* __restrict__ in, _Float16* __restrict__ out, int n4) {
    int i = blockIdx.x * 256 + threadIdx.x;
    if (i < n4) {
        float4 x = ((const float4*)in)[i];
        v4h h = { (_Float16)x.x, (_Float16)x.y, (_Float16)x.z, (_Float16)x.w };
        ((v4h*)out)[i] = h;
    }
}

// ---------------------------------------------------------------------------
// Pre-pass B: f32 [R,C] -> f16 [C,R] tiled transpose-convert through LDS
// ---------------------------------------------------------------------------
__global__ __launch_bounds__(256)
void transpose_cvt(const float* __restrict__ in, _Float16* __restrict__ out,
                   int R, int C, size_t inBatch, size_t outBatch)
{
    __shared__ __align__(16) _Float16 st[64 * LROW];   // transposed tile: st[c][r]
    const float* ib = in + (size_t)blockIdx.z * inBatch;
    _Float16* ob = out + (size_t)blockIdx.z * outBatch;
    const int c0 = blockIdx.x * 64, r0 = blockIdx.y * 64;
    const int tid = threadIdx.x;
    const int rr = tid >> 4;            // 0..15
    const int cc = (tid & 15) * 4;      // 0..60
    #pragma unroll
    for (int i = 0; i < 4; ++i) {
        int row = rr + i * 16;
        float4 x = *(const float4*)(ib + (size_t)(r0 + row) * C + c0 + cc);
        st[(cc + 0) * LROW + row] = (_Float16)x.x;
        st[(cc + 1) * LROW + row] = (_Float16)x.y;
        st[(cc + 2) * LROW + row] = (_Float16)x.z;
        st[(cc + 3) * LROW + row] = (_Float16)x.w;
    }
    __syncthreads();
    #pragma unroll
    for (int i = 0; i < 2; ++i) {       // 64x64 halves = 512 chunks of 8
        int c = tid + i * 256;
        int orow = c >> 3, osub = (c & 7) * 8;
        v8h v = *(const v8h*)(st + orow * LROW + osub);
        *(v8h*)(ob + (size_t)(c0 + orow) * R + r0 + osub) = v;
    }
}

// ---------------------------------------------------------------------------
// Pass 1: act[t, e*F+f] = f16( probs[t,e] * gelu(fc1) )
// grid = (F/128, T/128, E), block = 256
// ---------------------------------------------------------------------------
__global__ __launch_bounds__(256)
void moe_fc1_gelu_wmma(const _Float16* __restrict__ Xh,     // [T, H]
                       const float* __restrict__ probs,     // [T, E] (masked)
                       const _Float16* __restrict__ W1t,    // [E, F, H]
                       _Float16* __restrict__ act)          // [T, E*F]
{
    __shared__ __align__(16) _Float16 sA[2 * STAGE_HALVES];
    __shared__ __align__(16) _Float16 sB[2 * STAGE_HALVES];

    const int tid = threadIdx.x, wave = tid >> 5, lane = tid & 31;
    const int hs = lane >> 4, la = lane & 15;
    const int f0 = blockIdx.x * BNT;
    const int t0 = blockIdx.y * BM;
    const int e  = blockIdx.z;
    const int m0 = (wave >> 1) * 32;
    const int n0 = (wave & 1) * 64;

    v8f acc[2][4] = {};
    wmma_mainloop(Xh + (size_t)t0 * MOE_H,
                  W1t + ((size_t)e * MOE_F + f0) * MOE_H,
                  MOE_H, sA, sB, tid, lane, m0, n0, acc);

    #pragma unroll
    for (int i = 0; i < 2; ++i) {
        #pragma unroll
        for (int j = 0; j < 4; ++j) {
            v8f c = acc[i][j];
            const int f = f0 + n0 + j * 16 + la;
            #pragma unroll
            for (int v = 0; v < 8; ++v) {
                const int t = t0 + m0 + i * 16 + v + 8 * hs;
                float g  = gelu_exact(c[v]);
                float pv = probs[t * MOE_E + e];
                act[(size_t)t * MOE_K2 + e * MOE_F + f] = (_Float16)(g * pv);
            }
        }
    }
}

// ---------------------------------------------------------------------------
// Pass 2: out[t,h] = act[t,:] . W2t[h,:] + residual[t,h]
// grid = (H/128, T/128), block = 256
// ---------------------------------------------------------------------------
__global__ __launch_bounds__(256)
void moe_fc2_combine_wmma(const _Float16* __restrict__ act,  // [T, 16384]
                          const _Float16* __restrict__ W2t,  // [H, 16384]
                          const float* __restrict__ res,     // [T, H]
                          float* __restrict__ out)           // [T, H]
{
    __shared__ __align__(16) _Float16 sA[2 * STAGE_HALVES];
    __shared__ __align__(16) _Float16 sB[2 * STAGE_HALVES];

    const int tid = threadIdx.x, wave = tid >> 5, lane = tid & 31;
    const int hs = lane >> 4, la = lane & 15;
    const int h0 = blockIdx.x * BNT;
    const int t0 = blockIdx.y * BM;
    const int m0 = (wave >> 1) * 32;
    const int n0 = (wave & 1) * 64;

    v8f acc[2][4] = {};
    wmma_mainloop(act + (size_t)t0 * MOE_K2,
                  W2t + (size_t)h0 * MOE_K2,
                  MOE_K2, sA, sB, tid, lane, m0, n0, acc);

    #pragma unroll
    for (int i = 0; i < 2; ++i) {
        #pragma unroll
        for (int j = 0; j < 4; ++j) {
            v8f c = acc[i][j];
            const int h = h0 + n0 + j * 16 + la;
            #pragma unroll
            for (int v = 0; v < 8; ++v) {
                const int t = t0 + m0 + i * 16 + v + 8 * hs;
                out[(size_t)t * MOE_H + h] = c[v] + res[(size_t)t * MOE_H + h];
            }
        }
    }
}

// ---------------------------------------------------------------------------
extern "C" void kernel_launch(void* const* d_in, const int* in_sizes, int n_in,
                              void* d_out, int out_size, void* d_ws, size_t ws_size,
                              hipStream_t stream) {
    const float* X     = (const float*)d_in[0];  // [T,H]
    const float* res   = (const float*)d_in[1];  // [T,H]
    const float* probs = (const float*)d_in[2];  // [T,E] already masked
    // d_in[3] routing_map redundant (probs already masked) -> unused
    const float* W1    = (const float*)d_in[4];  // [E,H,F]
    const float* W2    = (const float*)d_in[5];  // [E,F,H] == [K2,H] flat
    float* out = (float*)d_out;

    // Workspace layout (halves): act | Xh | W1t | W2t   (total 132 MiB)
    _Float16* act = (_Float16*)d_ws;
    _Float16* Xh  = act + (size_t)MOE_T * MOE_K2;
    _Float16* W1t = Xh  + (size_t)MOE_T * MOE_H;
    _Float16* W2t = W1t + (size_t)MOE_E * MOE_F * MOE_H;

    dim3 blk(256, 1, 1);

    // Pre-pass: convert X; transpose-convert W1 ([E,H,F]->[E,F,H]) and W2 ([K2,H]->[H,K2]).
    cvt_f32_f16<<<dim3((MOE_T * MOE_H / 4 + 255) / 256), blk, 0, stream>>>(X, Xh, MOE_T * MOE_H / 4);
    transpose_cvt<<<dim3(MOE_F / 64, MOE_H / 64, MOE_E), blk, 0, stream>>>(
        W1, W1t, MOE_H, MOE_F, (size_t)MOE_H * MOE_F, (size_t)MOE_F * MOE_H);
    transpose_cvt<<<dim3(MOE_H / 64, MOE_K2 / 64, 1), blk, 0, stream>>>(
        W2, W2t, MOE_K2, MOE_H, 0, 0);

    dim3 g1(MOE_F / BNT, MOE_T / BM, MOE_E);     // 32 x 16 x 4
    dim3 g2(MOE_H / BNT, MOE_T / BM);            // 8 x 16
    moe_fc1_gelu_wmma<<<g1, blk, 0, stream>>>(Xh, probs, W1t, act);
    moe_fc2_combine_wmma<<<g2, blk, 0, stream>>>(act, W2t, res, out);

    (void)in_sizes; (void)n_in; (void)out_size; (void)ws_size;
}
